// ReactionAblationModel_26018911879336
// MI455X (gfx1250) — matrix operand order repaired and checked by reading, not compiled, not run on hardware
//
#include <hip/hip_runtime.h>
#include <hip/hip_bf16.h>
#include <math.h>

// ---------------------------------------------------------------------------
// Problem constants (match reference). All GEMM M dims are multiples of 32.
// ---------------------------------------------------------------------------
#define N_NODES 60000
#define N_EDGES 240000
#define N_GRAPH 1024
#define NHEAD   4
#define DH      32
#define HD      128   // NHEAD * DH
#define FN      17
#define FE      15

typedef __attribute__((ext_vector_type(16))) __bf16 v16bf;
typedef __attribute__((ext_vector_type(8)))  float  v8f;

// fp32 -> bf16 bits, round-to-nearest-even
__device__ __forceinline__ unsigned short bfbits(float f) {
  union { float f; unsigned u; } x; x.f = f;
  unsigned r = x.u + 0x7FFFu + ((x.u >> 16) & 1u);
  return (unsigned short)(r >> 16);
}

union Frag { v16bf v; uint4 q[2]; };

// ---------------------------------------------------------------------------
// Weight pack: W [K,Nc] fp32 row-major -> bf16 WMMA B-fragment order.
// Wp[((ks*(Nc/16) + ntile)*32 + lane)*16 + j], zero-padded past K.
// ---------------------------------------------------------------------------
__global__ void pack_w_kernel(const float* __restrict__ W,
                              unsigned short* __restrict__ Wp,
                              int K, int Nc) {
  int t = blockIdx.x * blockDim.x + threadIdx.x;
  const int ntt = Nc >> 4;
  const int Kp = (K + 31) >> 5;
  if (t >= Kp * ntt * 32) return;
  int lane = t & 31;
  int nt   = (t >> 5) % ntt;
  int ks   = (t >> 5) / ntt;
  int hi   = lane >> 4;
  int col  = nt * 16 + (lane & 15);
  unsigned short* dst = Wp + (size_t)t * 16;
  #pragma unroll
  for (int j = 0; j < 16; ++j) {
    int k = ks * 32 + hi * 16 + j;
    dst[j] = (k < K) ? bfbits(W[(size_t)k * Nc + col]) : (unsigned short)0;
  }
}

// ---------------------------------------------------------------------------
// bf16-WMMA GEMM: Y[M x Nc] = act( (A - A2?) @ W + bias ), M % 32 == 0,
// Nc % 128 == 0. Block = 256 thr = 8 waves covering a 32x128 output band.
// A: 32x32 fp32 slab -> bf16 in LDS per K-step (one float4 load + one packed
// ds_store_b64 per thread). Each wave: 1 B fragment (2x global_load_b128 from
// pre-packed weights), 2 A fragments (4x ds_load_b128), 2 WMMAs per K-step.
//  HAS_A2: A := A - A2;  SCATTER: Y[sidx[row]*ldy+col] += v;  TAIL: K%32 != 0
// ---------------------------------------------------------------------------
template<bool HAS_A2, bool SCATTER, int ACT, bool TAIL>
__global__ __launch_bounds__(256)
void gemm_wmma(const float* __restrict__ A,
               const float* __restrict__ A2,
               const unsigned short* __restrict__ Wp,
               const float* __restrict__ bias,
               float* __restrict__ Y,
               const int* __restrict__ sidx,
               int M, int Nc, int K, int ldy) {
  __shared__ unsigned short As[32 * 32];          // 32x32 bf16 A slab (2 KB)

  const int lane  = threadIdx.x & 31;
  const int wid   = threadIdx.x >> 5;
  const int nbn   = Nc >> 7;                      // 128-col blocks along N
  const int m0    = ((int)blockIdx.x / nbn) << 5; // 32-row band
  const int ntile = ((int)blockIdx.x % nbn) * 8 + wid;
  const int Kp    = (K + 31) >> 5;
  const int ntt   = Nc >> 4;

  // cooperative loader: 1024 elems, 4 per thread (one float4)
  const int lrow = (int)threadIdx.x >> 3;          // 0..31
  const int lcol = ((int)threadIdx.x & 7) * 4;     // 0,4,..,28
  const float* Arow  = A + (size_t)(m0 + lrow) * K;
  const float* A2row = HAS_A2 ? A2 + (size_t)(m0 + lrow) * K : (const float*)0;

  const int hi   = lane >> 4;
  const int bcol = ntile * 16 + (lane & 15);
  const uint4* aq0 = (const uint4*)&As[(lane & 15) * 32];        // rows 0..15
  const uint4* aq1 = (const uint4*)&As[(16 + (lane & 15)) * 32]; // rows 16..31

  v8f c0 = {}, c1 = {};
  for (int ks = 0; ks < Kp; ++ks) {
    // ---- stage A slab (fp32 -> bf16) into LDS ----
    {
      const int k = ks * 32 + lcol;
      float v0, v1, v2, v3;
      if (!TAIL) {                                // K % 32 == 0: no guards
        float4 x = *(const float4*)(Arow + k);
        v0 = x.x; v1 = x.y; v2 = x.z; v3 = x.w;
        if (HAS_A2) {
          float4 y = *(const float4*)(A2row + k);
          v0 -= y.x; v1 -= y.y; v2 -= y.z; v3 -= y.w;
        }
      } else {                                    // only K=17/15 layers
        v0 = (k     < K) ? Arow[k]     : 0.0f;
        v1 = (k + 1 < K) ? Arow[k + 1] : 0.0f;
        v2 = (k + 2 < K) ? Arow[k + 2] : 0.0f;
        v3 = (k + 3 < K) ? Arow[k + 3] : 0.0f;
      }
      uint2 d;
      d.x = (unsigned)bfbits(v0) | ((unsigned)bfbits(v1) << 16);
      d.y = (unsigned)bfbits(v2) | ((unsigned)bfbits(v3) << 16);
      *(uint2*)&As[lrow * 32 + lcol] = d;
    }
    __syncthreads();

    // ---- fragments ----
    Frag fb;                                      // packed weights: 32B/lane
    const uint4* bq = (const uint4*)(Wp + ((size_t)(ks * ntt + ntile) * 32 + lane) * 16);
    fb.q[0] = bq[0];
    fb.q[1] = bq[1];
    Frag fa0, fa1;                                // lanes 0-15: K={0..15}
    fa0.q[0] = aq0[2 * hi];  fa0.q[1] = aq0[2 * hi + 1];  // lanes 16-31: K={16..31}
    fa1.q[0] = aq1[2 * hi];  fa1.q[1] = aq1[2 * hi + 1];

    c0 = __builtin_amdgcn_wmma_f32_16x16x32_bf16(false, fa0.v, false, fb.v,
                                                 (short)0, c0, false, false);
    c1 = __builtin_amdgcn_wmma_f32_16x16x32_bf16(false, fa1.v, false, fb.v,
                                                 (short)0, c1, false, false);
    __syncthreads();
  }

  const float bv = bias[bcol];
  #pragma unroll
  for (int v = 0; v < 8; ++v) {
    int row0 = m0 + v + hi * 8;                   // tile 0: rows m0..m0+15
    int row1 = row0 + 16;                         // tile 1: rows m0+16..m0+31
    float f0 = c0[v] + bv;
    float f1 = c1[v] + bv;
    if (ACT == 1) { f0 = f0 > 0.0f ? f0 : 0.0f; f1 = f1 > 0.0f ? f1 : 0.0f; }
    if (SCATTER) {
      atomicAdd(Y + (size_t)sidx[row0] * ldy + bcol, f0);
      atomicAdd(Y + (size_t)sidx[row1] * ldy + bcol, f1);
    } else {
      Y[(size_t)row0 * ldy + bcol] = f0;
      Y[(size_t)row1 * ldy + bcol] = f1;
    }
  }
}

// ---------------------------------------------------------------------------
// Elementwise / scatter kernels for the EGAT layer
// ---------------------------------------------------------------------------
__global__ void fill_kernel(float* __restrict__ p, float v, int n) {
  int i = blockIdx.x * blockDim.x + threadIdx.x;
  if (i < n) p[i] = v;
}

// f_out = leaky_relu(eW + f_ni[src] + f_nj[dst]); logits = <f_out, attn>; seg-max by dst
__global__ void edge_combine_kernel(float* __restrict__ f_out,
                                    const float* __restrict__ f_ni,
                                    const float* __restrict__ f_nj,
                                    const int* __restrict__ src,
                                    const int* __restrict__ dst,
                                    const float* __restrict__ attn,
                                    float* __restrict__ logits,
                                    float* __restrict__ segmax) {
  int t = blockIdx.x * blockDim.x + threadIdx.x;
  if (t >= N_EDGES * NHEAD) return;
  int e = t >> 2, h = t & 3;
  int s = src[e], d = dst[e];
  const float* pi = f_ni + (size_t)s * HD + h * DH;
  const float* pj = f_nj + (size_t)d * HD + h * DH;
  const float* pa = attn + h * DH;
  float* pe = f_out + (size_t)e * HD + h * DH;
  float acc = 0.0f;
  #pragma unroll
  for (int k = 0; k < DH; ++k) {
    float v = pe[k] + pi[k] + pj[k];
    v = v > 0.0f ? v : 0.01f * v;
    pe[k] = v;
    acc += v * pa[k];
  }
  logits[t] = acc;
  float* ma = segmax + (size_t)d * NHEAD + h;     // float atomic-max, sign-split
  if (acc >= 0.0f) atomicMax((int*)ma, __float_as_int(acc));
  else             atomicMin((unsigned int*)ma, __float_as_uint(acc));
}

__global__ void edge_softmax_kernel(float* __restrict__ logits,
                                    const int* __restrict__ dst,
                                    const float* __restrict__ segmax,
                                    float* __restrict__ segsum) {
  int t = blockIdx.x * blockDim.x + threadIdx.x;
  if (t >= N_EDGES * NHEAD) return;
  int e = t >> 2, h = t & 3;
  int d = dst[e];
  float ex = __expf(logits[t] - segmax[(size_t)d * NHEAD + h]);
  logits[t] = ex;
  atomicAdd(&segsum[(size_t)d * NHEAD + h], ex);
}

__global__ void message_kernel(const float* __restrict__ logits,
                               const float* __restrict__ segsum,
                               const float* __restrict__ h_node,
                               const int* __restrict__ src,
                               const int* __restrict__ dst,
                               float* __restrict__ h_out) {
  int t = blockIdx.x * blockDim.x + threadIdx.x;
  if (t >= N_EDGES * NHEAD) return;
  int e = t >> 2, h = t & 3;
  int s = src[e], d = dst[e];
  float a = logits[t] / segsum[(size_t)d * NHEAD + h];
  const float* pn = h_node + (size_t)s * HD + h * DH;
  float* po = h_out + (size_t)d * HD + h * DH;
  #pragma unroll
  for (int k = 0; k < DH; ++k) atomicAdd(&po[k], pn[k] * a);
}

__global__ void mlp_out_kernel(const float* __restrict__ x2,
                               const float* __restrict__ Wm3,
                               const float* __restrict__ bm3,
                               float* __restrict__ out) {
  int g = blockIdx.x * blockDim.x + threadIdx.x;
  if (g >= N_GRAPH) return;
  const float* p = x2 + (size_t)g * 256;
  float acc = bm3[0];
  for (int k = 0; k < 256; ++k) acc += p[k] * Wm3[k];
  out[g] = acc;
}

// ---------------------------------------------------------------------------
// Host-side orchestration
// ---------------------------------------------------------------------------
struct EgatW {
  const float *bni, *bnj, *bf, *bnode, *attn;
  const unsigned short *Wni, *Wnj, *Wf, *Wnode;
};

static inline void launch_pack(hipStream_t s, const float* W, unsigned short* Wp,
                               int K, int Nc) {
  int total = ((K + 31) / 32) * (Nc / 16) * 32;
  pack_w_kernel<<<dim3((total + 255) / 256), dim3(256), 0, s>>>(W, Wp, K, Nc);
}

static inline void launch_gemm(hipStream_t s, const float* A, const float* A2,
                               const unsigned short* Wp, const float* b, float* Y,
                               const int* sidx, int M, int Nc, int K, int ldy, int act) {
  dim3 grid((M / 32) * (Nc / 128)), blk(256);
  const bool tail = (K & 31) != 0;
  if (A2)        gemm_wmma<true,  true,  0, false><<<grid, blk, 0, s>>>(A, A2, Wp, b, Y, sidx, M, Nc, K, ldy);
  else if (act)  gemm_wmma<false, false, 1, false><<<grid, blk, 0, s>>>(A, A2, Wp, b, Y, sidx, M, Nc, K, ldy);
  else if (tail) gemm_wmma<false, false, 0, true ><<<grid, blk, 0, s>>>(A, A2, Wp, b, Y, sidx, M, Nc, K, ldy);
  else           gemm_wmma<false, false, 0, false><<<grid, blk, 0, s>>>(A, A2, Wp, b, Y, sidx, M, Nc, K, ldy);
}

static inline void launch_fill(hipStream_t s, float* p, float v, int n) {
  fill_kernel<<<dim3((n + 255) / 256), dim3(256), 0, s>>>(p, v, n);
}

static void run_egat(hipStream_t s,
                     const float* h_in, int Kn, const float* e_in, int Ke,
                     const int* src, const int* dst, const EgatW& w,
                     float* nA, float* nB, float* nC,
                     float* logits, float* smax, float* ssum,
                     float* h_out, float* e_out) {
  const int ehb = (N_EDGES * NHEAD + 255) / 256;
  launch_gemm(s, h_in, nullptr, w.Wni,   w.bni,   nA,    nullptr, N_NODES, HD, Kn, HD, 0);
  launch_gemm(s, h_in, nullptr, w.Wnj,   w.bnj,   nB,    nullptr, N_NODES, HD, Kn, HD, 0);
  launch_gemm(s, h_in, nullptr, w.Wnode, w.bnode, nC,    nullptr, N_NODES, HD, Kn, HD, 0);
  launch_gemm(s, e_in, nullptr, w.Wf,    w.bf,    e_out, nullptr, N_EDGES, HD, Ke, HD, 0);
  launch_fill(s, smax, -INFINITY, N_NODES * NHEAD);
  launch_fill(s, ssum, 0.0f,      N_NODES * NHEAD);
  launch_fill(s, h_out, 0.0f,     N_NODES * HD);
  edge_combine_kernel<<<dim3(ehb), dim3(256), 0, s>>>(e_out, nA, nB, src, dst, w.attn, logits, smax);
  edge_softmax_kernel<<<dim3(ehb), dim3(256), 0, s>>>(logits, dst, smax, ssum);
  message_kernel<<<dim3(ehb), dim3(256), 0, s>>>(logits, ssum, nC, src, dst, h_out);
}

extern "C" void kernel_launch(void* const* d_in, const int* in_sizes, int n_in,
                              void* d_out, int out_size, void* d_ws, size_t ws_size,
                              hipStream_t stream) {
  (void)in_sizes; (void)n_in; (void)out_size; (void)ws_size;
  const float* hR  = (const float*)d_in[0];
  const float* eR  = (const float*)d_in[1];
  const float* hP  = (const float*)d_in[2];
  const float* eP  = (const float*)d_in[3];
  const int* srcR  = (const int*)d_in[4];
  const int* dstR  = (const int*)d_in[5];
  const int* srcP  = (const int*)d_in[6];
  const int* dstP  = (const int*)d_in[7];
  const int* nid   = (const int*)d_in[8];
  const int* eid   = (const int*)d_in[9];
  float* out = (float*)d_out;

  // ---- workspace carve: fp32 activations, then packed bf16 weights ----
  float* p = (float*)d_ws;
  auto take = [&](size_t n) { float* r = p; p += n; return r; };
  float* nA     = take((size_t)N_NODES * HD);
  float* nB     = take((size_t)N_NODES * HD);
  float* nC     = take((size_t)N_NODES * HD);
  float* h0     = take((size_t)N_NODES * HD);
  float* h1     = take((size_t)N_NODES * HD);
  float* hPb    = take((size_t)N_NODES * HD);
  float* e0     = take((size_t)N_EDGES * HD);
  float* e1     = take((size_t)N_EDGES * HD);
  float* ePb    = take((size_t)N_EDGES * HD);
  float* logits = take((size_t)N_EDGES * NHEAD);
  float* smax   = take((size_t)N_NODES * NHEAD);
  float* ssum   = take((size_t)N_NODES * NHEAD);
  float* Gf     = take((size_t)N_GRAPH * 256);
  float* x1     = take((size_t)N_GRAPH * 512);
  float* x2     = take((size_t)N_GRAPH * 256);

  unsigned short* wp = (unsigned short*)p;
  auto takew = [&](int K, int Nc) {
    unsigned short* r = wp; wp += (size_t)((K + 31) / 32) * Nc * 32; return r;
  };
  unsigned short* p1ni = takew(FN, HD);  unsigned short* p1nj = takew(FN, HD);
  unsigned short* p1f  = takew(FE, HD);  unsigned short* p1no = takew(FN, HD);
  unsigned short* p2ni = takew(HD, HD);  unsigned short* p2nj = takew(HD, HD);
  unsigned short* p2f  = takew(HD, HD);  unsigned short* p2no = takew(HD, HD);
  unsigned short* pan  = takew(HD, HD);  unsigned short* pae  = takew(HD, HD);
  unsigned short* pm1  = takew(256, 512);
  unsigned short* pm2  = takew(512, 256);

  // ---- pack all weights (every call: deterministic, capture-safe) ----
  launch_pack(stream, (const float*)d_in[10], p1ni, FN, HD);
  launch_pack(stream, (const float*)d_in[12], p1nj, FN, HD);
  launch_pack(stream, (const float*)d_in[14], p1f,  FE, HD);
  launch_pack(stream, (const float*)d_in[16], p1no, FN, HD);
  launch_pack(stream, (const float*)d_in[19], p2ni, HD, HD);
  launch_pack(stream, (const float*)d_in[21], p2nj, HD, HD);
  launch_pack(stream, (const float*)d_in[23], p2f,  HD, HD);
  launch_pack(stream, (const float*)d_in[25], p2no, HD, HD);
  launch_pack(stream, (const float*)d_in[28], pan,  HD, HD);
  launch_pack(stream, (const float*)d_in[30], pae,  HD, HD);
  launch_pack(stream, (const float*)d_in[32], pm1,  256, 512);
  launch_pack(stream, (const float*)d_in[34], pm2,  512, 256);

  EgatW w1 { (const float*)d_in[11], (const float*)d_in[13], (const float*)d_in[15],
             (const float*)d_in[17], (const float*)d_in[18], p1ni, p1nj, p1f, p1no };
  EgatW w2 { (const float*)d_in[20], (const float*)d_in[22], (const float*)d_in[24],
             (const float*)d_in[26], (const float*)d_in[27], p2ni, p2nj, p2f, p2no };

  // ---- P branch: egat1 only ----
  run_egat(stream, hP, FN, eP, FE, srcP, dstP, w1,
           nA, nB, nC, logits, smax, ssum, hPb, ePb);

  // ---- R branch: egat1 + 2x egat2 ----
  run_egat(stream, hR, FN, eR, FE, srcR, dstR, w1,
           nA, nB, nC, logits, smax, ssum, h0, e0);
  run_egat(stream, h0, HD, e0, HD, srcR, dstR, w2,
           nA, nB, nC, logits, smax, ssum, h1, e1);
  run_egat(stream, h1, HD, e1, HD, srcR, dstR, w2,
           nA, nB, nC, logits, smax, ssum, h0, e0);

  // ---- Readout: diff fused into A-load, segment_sum fused into epilogue ----
  launch_fill(stream, Gf, 0.0f, N_GRAPH * 256);
  launch_gemm(stream, hPb, h0, pan, (const float*)d_in[29], Gf,      nid, N_NODES, HD, HD, 256, 0);
  launch_gemm(stream, ePb, e0, pae, (const float*)d_in[31], Gf + HD, eid, N_EDGES, HD, HD, 256, 0);

  // ---- MLP head ----
  launch_gemm(stream, Gf, nullptr, pm1, (const float*)d_in[33], x1, nullptr, N_GRAPH, 512, 256, 512, 1);
  launch_gemm(stream, x1, nullptr, pm2, (const float*)d_in[35], x2, nullptr, N_GRAPH, 256, 512, 256, 1);
  mlp_out_kernel<<<dim3((N_GRAPH + 255) / 256), dim3(256), 0, stream>>>(
      x2, (const float*)d_in[36], (const float*)d_in[37], out);
}
